// DeformableTransformerDecoderLayer_46875273069198
// MI455X (gfx1250) — matrix operand order repaired and checked by reading, not compiled, not run on hardware
//
#include <hip/hip_runtime.h>
#include <math.h>

typedef __attribute__((ext_vector_type(16))) _Float16 v16h;
typedef __attribute__((ext_vector_type(8)))  float    v8f;
typedef __attribute__((ext_vector_type(4)))  unsigned int u32x4;
typedef __attribute__((ext_vector_type(8)))  int      i32x8;
typedef __attribute__((ext_vector_type(4)))  int      i32x4;

static constexpr int  kB  = 8;
static constexpr int  kPQ = 900;
static constexpr int  kC  = 256;
static constexpr int  kNH = 8;
static constexpr int  kFF = 2048;
static constexpr int  kPK = 20197;
static constexpr int  kMq = kPQ * kB;          // 7200 rows, ordering (s*B + b)
static constexpr long kMv = (long)kB * kPK;    // 161576 rows, ordering (b*PK + p)

__device__ __forceinline__ v8f wmma16(v16h a, v16h b, v8f c) {
  return __builtin_amdgcn_wmma_f32_16x16x32_f16(false, a, false, b, (short)0, c, false, false);
}

// ---------------------------------------------------------------------------
// TDM: async 2D tile load (global -> LDS), descriptor per CDNA5 ISA ch.8.
// tile: tile1 rows x tile0 elements, row stride = `stride` elements.
// Rows/cols beyond tensor_dim are zero-filled by the TDM.
// dsz_code: 0=1B,1=2B,2=4B,3=8B elements.
// ---------------------------------------------------------------------------
__device__ __forceinline__ void tdm_load_2d(unsigned lds_addr, const void* gptr,
                                            unsigned dim0, unsigned dim1,
                                            unsigned tile0, unsigned tile1,
                                            unsigned long stride, int dsz_code) {
  unsigned long ga = (unsigned long)(uintptr_t)gptr;
  u32x4 g0;
  g0[0] = 1u;                                            // count=1 (valid), user mode
  g0[1] = lds_addr;                                      // LDS byte address
  g0[2] = (unsigned)(ga & 0xffffffffu);                  // global addr [31:0]
  g0[3] = (unsigned)((ga >> 32) & 0x1ffffffu) | (2u << 30);  // addr[56:32] | type=2
  i32x8 g1;
  g1[0] = (int)((unsigned)dsz_code << 16);               // wg_mask=0 | data_size
  g1[1] = (int)((dim0 & 0xffffu) << 16);                 // tensor_dim0[15:0]
  g1[2] = (int)((dim0 >> 16) | ((dim1 & 0xffffu) << 16));// dim0[31:16] | dim1[15:0]
  g1[3] = (int)((dim1 >> 16) | (tile0 << 16));           // dim1[31:16] | tile_dim0
  g1[4] = (int)(tile1 & 0xffffu);                        // tile_dim1 | tile_dim2=0
  g1[5] = (int)(stride & 0xffffffffu);                   // tensor_dim0_stride[31:0]
  g1[6] = (int)((stride >> 32) & 0xffffu);               // stride[47:32] | dim1_stride lo=0
  g1[7] = 0;
  i32x4 z4 = {0, 0, 0, 0};
  i32x8 z8 = {0, 0, 0, 0, 0, 0, 0, 0};
  __builtin_amdgcn_tensor_load_to_lds(g0, g1, z4, z4, z8, 0);
}

// ---------------------------------------------------------------------------
// Layout transposes
// ---------------------------------------------------------------------------
__global__ __launch_bounds__(256) void make_xq(const float* __restrict__ q, float* __restrict__ xq) {
  int row = blockIdx.x, c = threadIdx.x;          // row = s*8 + b
  int s = row >> 3, b = row & 7;
  xq[(size_t)row * kC + c] = q[((size_t)b * kC + c) * kPQ + s];
}

__global__ __launch_bounds__(256) void make_kf(const float* __restrict__ k, _Float16* __restrict__ kf) {
  long row = blockIdx.x; int c = threadIdx.x;     // row = b*PK + p
  long b = row / kPK, p = row - b * kPK;
  kf[(size_t)row * kC + c] = (_Float16)k[((size_t)b * kC + c) * kPK + p];
}

__global__ __launch_bounds__(256) void make_out(const float* __restrict__ q3, float* __restrict__ out) {
  int row = blockIdx.x, c = threadIdx.x;
  int s = row >> 3, b = row & 7;
  out[((size_t)b * kC + c) * kPQ + s] = q3[(size_t)row * kC + c];
}

// ---------------------------------------------------------------------------
// WMMA GEMM with TDM double-buffered LDS staging.
// Y[M,N] = act(X[M,K] (f32 or f16) @ W[N,K]^T + bias)
// Block = 8 waves -> 64(M) x 128(N); wave -> 32x32 (2x2 WMMA frags).
// Wave 0 DMAs the next K-step's A (64x32) and B (128x32) tiles into LDS with
// tensor_load_to_lds while all waves compute from the resident buffer.
// Requires: K % 32 == 0, N % 128 == 0. M handled by TDM zero-fill + store guard.
// ---------------------------------------------------------------------------
template <typename TX>
__global__ __launch_bounds__(256) void gemm_xwT(const TX* __restrict__ X,
                                                const float* __restrict__ W,
                                                const float* __restrict__ bias,
                                                float* __restrict__ Yf,
                                                _Float16* __restrict__ Yh,
                                                long M, int N, int K, int relu) {
  __shared__ TX    As[2][64 * 32];
  __shared__ float Bs[2][128 * 32];
  const int lane = threadIdx.x & 31;
  const int wave = threadIdx.x >> 5;
  const int wm = wave >> 2, wn = wave & 3;
  const long mblk = (long)blockIdx.y * 64;
  const int  nblk = blockIdx.x * 128;
  const int c = lane & 15, hi = lane >> 4;
  const int dszX = (sizeof(TX) == 2) ? 1 : 2;

  unsigned dim1A = (unsigned)(M - mblk);   // remaining rows; TDM zero-fills past this

  auto issue = [&](int bi, int k0) {
    tdm_load_2d((unsigned)(uintptr_t)&As[bi][0], X + mblk * K + k0,
                (unsigned)(K - k0), dim1A, 32u, 64u, (unsigned long)K, dszX);
    tdm_load_2d((unsigned)(uintptr_t)&Bs[bi][0], W + (size_t)nblk * K + k0,
                (unsigned)(K - k0), (unsigned)(N - nblk), 32u, 128u,
                (unsigned long)K, 2);
  };

  if (threadIdx.x < 32) issue(0, 0);

  v8f a00 = {}, a01 = {}, a10 = {}, a11 = {};
  const int rl0 = wm * 32 + c, rl1 = wm * 32 + 16 + c;
  const int nl0 = wn * 32 + c, nl1 = wn * 32 + 16 + c;

  for (int k0 = 0; k0 < K; k0 += 32) {
    const int bi = (k0 >> 5) & 1;
    if (threadIdx.x < 32) {
      if (k0 + 32 < K) issue(bi ^ 1, k0 + 32);
      // TDM ops of one wave complete in order: <=2 pending leaves only the
      // prefetched (next-buffer) pair outstanding.
      if (k0 + 32 < K) __builtin_amdgcn_s_wait_tensorcnt((short)2);
      else             __builtin_amdgcn_s_wait_tensorcnt((short)0);
    }
    __syncthreads();

    const TX*    a0p = &As[bi][rl0 * 32];
    const TX*    a1p = &As[bi][rl1 * 32];
    const float* b0p = &Bs[bi][nl0 * 32];
    const float* b1p = &Bs[bi][nl1 * 32];
    v16h fa0, fa1, fb0, fb1;
#pragma unroll
    for (int e = 0; e < 8; ++e) {
      fa0[e]     = (_Float16)(float)a0p[hi * 8 + e];
      fa0[8 + e] = (_Float16)(float)a0p[16 + hi * 8 + e];
      fa1[e]     = (_Float16)(float)a1p[hi * 8 + e];
      fa1[8 + e] = (_Float16)(float)a1p[16 + hi * 8 + e];
    }
#pragma unroll
    for (int e = 0; e < 8; ++e) {
      fb0[e]     = (_Float16)b0p[hi * 8 + e];
      fb0[8 + e] = (_Float16)b0p[16 + hi * 8 + e];
      fb1[e]     = (_Float16)b1p[hi * 8 + e];
      fb1[8 + e] = (_Float16)b1p[16 + hi * 8 + e];
    }
    a00 = wmma16(fa0, fb0, a00);
    a01 = wmma16(fa0, fb1, a01);
    a10 = wmma16(fa1, fb0, a10);
    a11 = wmma16(fa1, fb1, a11);
    __syncthreads();   // all waves done reading buffer bi before it is re-filled
  }

  const long m0 = mblk + wm * 32;
  const int  n0 = nblk + wn * 32;
  auto stor = [&](v8f& acc, long mt, int nt) {
    int n = nt + c;
    float bv = bias[n];
#pragma unroll
    for (int v = 0; v < 8; ++v) {
      long m = mt + v + 8 * hi;
      if (m < M) {
        float t = acc[v] + bv;
        if (relu) t = fmaxf(t, 0.f);
        if (Yh) Yh[m * N + n] = (_Float16)t;
        else    Yf[m * N + n] = t;
      }
    }
  };
  stor(a00, m0, n0);      stor(a01, m0, n0 + 16);
  stor(a10, m0 + 16, n0); stor(a11, m0 + 16, n0 + 16);
}

// ---------------------------------------------------------------------------
// Flash self-attention over QKV (rows s*8+b, 768 cols = q|k|v, head h = 32 dims).
// One wave per (16-query tile, b, h). Online softmax, WMMA for QK^T and PV.
// ---------------------------------------------------------------------------
__global__ __launch_bounds__(32) void flash_attn(const float* __restrict__ QKV,
                                                 float* __restrict__ O) {
  const int lane = threadIdx.x & 31;
  const int c = lane & 15, hi = lane >> 4;
  const int s0 = blockIdx.x * 16;
  const int b = blockIdx.y >> 3, h = blockIdx.y & 7;
  const float scale = 0.17677669529663687f;  // 1/sqrt(32)

  v16h aQ;
  {
    int r = s0 + c; if (r > kPQ - 1) r = kPQ - 1;
    const float* qp = QKV + (size_t)(r * kB + b) * 768 + h * 32;
#pragma unroll
    for (int e = 0; e < 8; ++e) {
      aQ[e]     = (_Float16)(qp[hi * 8 + e] * scale);
      aQ[8 + e] = (_Float16)(qp[16 + hi * 8 + e] * scale);
    }
  }

  float mrow[8], lrow[8];
#pragma unroll
  for (int v = 0; v < 8; ++v) { mrow[v] = -1e30f; lrow[v] = 0.f; }
  v8f accO0 = {}, accO1 = {};
  __shared__ float Pl[16 * 32];

  for (int k0 = 0; k0 < kPQ; k0 += 32) {
    v8f sc[2];
#pragma unroll
    for (int j = 0; j < 2; ++j) {
      int key = k0 + j * 16 + c;
      int kc = key > kPQ - 1 ? kPQ - 1 : key;
      const float* kp = QKV + (size_t)(kc * kB + b) * 768 + 256 + h * 32;
      v16h bK;
#pragma unroll
      for (int e = 0; e < 8; ++e) { bK[e] = (_Float16)kp[hi * 8 + e]; bK[8 + e] = (_Float16)kp[16 + hi * 8 + e]; }
      v8f z = {};
      sc[j] = wmma16(aQ, bK, z);
      if (key >= kPQ) {
#pragma unroll
        for (int v = 0; v < 8; ++v) sc[j][v] = -1e30f;
      }
    }
    float alpha[8];
#pragma unroll
    for (int v = 0; v < 8; ++v) {
      float rm = fmaxf(sc[0][v], sc[1][v]);
#pragma unroll
      for (int o = 8; o >= 1; o >>= 1) rm = fmaxf(rm, __shfl_xor(rm, o, 32));
      float mn = fmaxf(mrow[v], rm);
      alpha[v] = __expf(mrow[v] - mn);
      mrow[v] = mn;
      sc[0][v] = __expf(sc[0][v] - mn);
      sc[1][v] = __expf(sc[1][v] - mn);
      float rs = sc[0][v] + sc[1][v];
#pragma unroll
      for (int o = 8; o >= 1; o >>= 1) rs += __shfl_xor(rs, o, 32);
      lrow[v] = lrow[v] * alpha[v] + rs;
      accO0[v] *= alpha[v];
      accO1[v] *= alpha[v];
    }
    __syncthreads();
#pragma unroll
    for (int j = 0; j < 2; ++j)
#pragma unroll
      for (int v = 0; v < 8; ++v)
        Pl[(v + 8 * hi) * 32 + j * 16 + c] = sc[j][v];
    __syncthreads();
    v16h aP;
#pragma unroll
    for (int e = 0; e < 8; ++e) {
      aP[e]     = (_Float16)Pl[c * 32 + hi * 8 + e];
      aP[8 + e] = (_Float16)Pl[c * 32 + 16 + hi * 8 + e];
    }
#pragma unroll
    for (int jd = 0; jd < 2; ++jd) {
      v16h bV;
      int d = jd * 16 + c;
#pragma unroll
      for (int e = 0; e < 16; ++e) {
        int key = k0 + (e < 8 ? hi * 8 + e : 16 + hi * 8 + (e - 8)); if (key > kPQ - 1) key = kPQ - 1;
        bV[e] = (_Float16)QKV[(size_t)(key * kB + b) * 768 + 512 + h * 32 + d];
      }
      if (jd == 0) accO0 = wmma16(aP, bV, accO0);
      else         accO1 = wmma16(aP, bV, accO1);
    }
  }
#pragma unroll
  for (int v = 0; v < 8; ++v) {
    int s = s0 + v + 8 * hi;
    if (s < kPQ) {
      float inv = 1.0f / lrow[v];
      O[(size_t)(s * kB + b) * kC + h * 32 + c]      = accO0[v] * inv;
      O[(size_t)(s * kB + b) * kC + h * 32 + 16 + c] = accO1[v] * inv;
    }
  }
}

// ---------------------------------------------------------------------------
// y = LayerNorm(x + r) over C=256, one block per row.
// ---------------------------------------------------------------------------
__global__ __launch_bounds__(256) void add_ln(const float* __restrict__ x, const float* __restrict__ r,
                                              const float* __restrict__ g, const float* __restrict__ b,
                                              float* __restrict__ y) {
  int row = blockIdx.x, c = threadIdx.x;
  float v = x[(size_t)row * kC + c] + r[(size_t)row * kC + c];
  float s = v, s2 = v * v;
#pragma unroll
  for (int o = 16; o >= 1; o >>= 1) { s += __shfl_xor(s, o, 32); s2 += __shfl_xor(s2, o, 32); }
  __shared__ float ls[8], ls2[8];
  int w = threadIdx.x >> 5;
  if ((threadIdx.x & 31) == 0) { ls[w] = s; ls2[w] = s2; }
  __syncthreads();
  if (threadIdx.x == 0) {
    float t = 0, t2 = 0;
#pragma unroll
    for (int i = 0; i < 8; ++i) { t += ls[i]; t2 += ls2[i]; }
    ls[0] = t; ls2[0] = t2;
  }
  __syncthreads();
  float mean = ls[0] * (1.0f / kC);
  float var = ls2[0] * (1.0f / kC) - mean * mean;
  y[(size_t)row * kC + c] = (v - mean) * rsqrtf(var + 1e-5f) * g[c] + b[c];
}

// softmax over 16 (L*P) per (row, head), in place. aw: [7200,128]
__global__ __launch_bounds__(256) void aw_softmax(float* __restrict__ aw) {
  int idx = blockIdx.x * blockDim.x + threadIdx.x;
  if (idx >= kMq * kNH) return;
  float* p = aw + (size_t)(idx >> 3) * 128 + (idx & 7) * 16;
  float mx = p[0];
#pragma unroll
  for (int i = 1; i < 16; ++i) mx = fmaxf(mx, p[i]);
  float e[16], s = 0.f;
#pragma unroll
  for (int i = 0; i < 16; ++i) { e[i] = __expf(p[i] - mx); s += e[i]; }
  float inv = 1.f / s;
#pragma unroll
  for (int i = 0; i < 16; ++i) p[i] = e[i] * inv;
}

// ---------------------------------------------------------------------------
// Multi-scale deformable sampling. Block = one (s,b) row; thread = h*32 + d.
// ---------------------------------------------------------------------------
__device__ __forceinline__ float dsamp(const _Float16* vb, int ix, int iy, int W, int H) {
  if (ix < 0 || ix >= W || iy < 0 || iy >= H) return 0.f;
  return (float)vb[(size_t)(iy * W + ix) * kC];
}

__global__ __launch_bounds__(256) void ms_deform_k(const _Float16* __restrict__ val,
                                                   const float* __restrict__ off,
                                                   const float* __restrict__ aw,
                                                   const float* __restrict__ refp,
                                                   float* __restrict__ out) {
  const int row = blockIdx.x;          // s*8 + b
  const int s = row >> 3, b = row & 7;
  const int h = threadIdx.x >> 5, d = threadIdx.x & 31;
  const int Hl[4] = {100, 50, 25, 13};
  const int Wl[4] = {152, 76, 38, 19};
  const int st[4] = {0, 15200, 19000, 19950};
  const float* orow = off + (size_t)row * 256;
  const float* arow = aw + (size_t)row * 128;
  float acc = 0.f;
#pragma unroll
  for (int l = 0; l < 4; ++l) {
    const float rx = refp[((size_t)(b * kPQ + s) * 4 + l) * 2 + 0];
    const float ry = refp[((size_t)(b * kPQ + s) * 4 + l) * 2 + 1];
    const float fW = (float)Wl[l], fH = (float)Hl[l];
    const _Float16* vb = val + ((size_t)(b * kPK + st[l])) * kC + h * 32 + d;
#pragma unroll
    for (int p = 0; p < 4; ++p) {
      int o = ((h * 4 + l) * 4 + p) * 2;
      float x = (rx + orow[o] / fW) * fW - 0.5f;
      float y = (ry + orow[o + 1] / fH) * fH - 0.5f;
      float x0f = floorf(x), y0f = floorf(y);
      float lw = x - x0f, lh = y - y0f;
      int x0 = (int)x0f, y0 = (int)y0f;
      float w = arow[(h * 4 + l) * 4 + p];
      float s00 = dsamp(vb, x0, y0, Wl[l], Hl[l]);
      float s10 = dsamp(vb, x0 + 1, y0, Wl[l], Hl[l]);
      float s01 = dsamp(vb, x0, y0 + 1, Wl[l], Hl[l]);
      float s11 = dsamp(vb, x0 + 1, y0 + 1, Wl[l], Hl[l]);
      acc += w * (s00 * (1.f - lw) * (1.f - lh) + s10 * lw * (1.f - lh) +
                  s01 * (1.f - lw) * lh + s11 * lw * lh);
    }
  }
  out[(size_t)row * kC + threadIdx.x] = acc;
}

// ---------------------------------------------------------------------------
static void gemm_f(const float* X, const float* W, const float* bias, float* Yf,
                   long M, int N, int K, int relu, hipStream_t s) {
  dim3 grid(N / 128, (unsigned)((M + 63) / 64));
  gemm_xwT<float><<<grid, 256, 0, s>>>(X, W, bias, Yf, nullptr, M, N, K, relu);
}
static void gemm_h(const _Float16* X, const float* W, const float* bias, _Float16* Yh,
                   long M, int N, int K, hipStream_t s) {
  dim3 grid(N / 128, (unsigned)((M + 63) / 64));
  gemm_xwT<_Float16><<<grid, 256, 0, s>>>(X, W, bias, nullptr, Yh, M, N, K, 0);
}

extern "C" void kernel_launch(void* const* d_in, const int* in_sizes, int n_in,
                              void* d_out, int out_size, void* d_ws, size_t ws_size,
                              hipStream_t stream) {
  (void)in_sizes; (void)n_in; (void)out_size; (void)ws_size;
  const float* query = (const float*)d_in[0];
  const float* key   = (const float*)d_in[1];
  const float* refp  = (const float*)d_in[4];
  const float* W_in  = (const float*)d_in[5];
  const float* b_in  = (const float*)d_in[6];
  const float* W_out = (const float*)d_in[7];
  const float* b_out = (const float*)d_in[8];
  const float* W_off = (const float*)d_in[9];
  const float* b_off = (const float*)d_in[10];
  const float* W_aw  = (const float*)d_in[11];
  const float* b_aw  = (const float*)d_in[12];
  const float* W_v   = (const float*)d_in[13];
  const float* b_v   = (const float*)d_in[14];
  const float* W_o   = (const float*)d_in[15];
  const float* b_o   = (const float*)d_in[16];
  const float* W1    = (const float*)d_in[17];
  const float* b1    = (const float*)d_in[18];
  const float* W2    = (const float*)d_in[19];
  const float* b2    = (const float*)d_in[20];
  const float* g1    = (const float*)d_in[21];
  const float* be1   = (const float*)d_in[22];
  const float* g2    = (const float*)d_in[23];
  const float* be2   = (const float*)d_in[24];
  const float* g3    = (const float*)d_in[25];
  const float* be3   = (const float*)d_in[26];

  char* wp = (char*)d_ws;
  auto alloc = [&](size_t bytes) -> void* {
    void* r = (void*)wp;
    wp += (bytes + 255) & ~(size_t)255;
    return r;
  };
  float*    xq   = (float*)alloc((size_t)kMq * kC * 4);        // residual stream / final q
  float*    qkv  = (float*)alloc((size_t)kMq * 3 * kC * 4);    // qkv, later ffn out
  float*    attn = (float*)alloc((size_t)kMq * kC * 4);        // flash out, later W_o out
  float*    q1   = (float*)alloc((size_t)kMq * kC * 4);
  float*    q2   = (float*)alloc((size_t)kMq * kC * 4);
  float*    offb = (float*)alloc((size_t)kMq * 256 * 4);
  float*    awb  = (float*)alloc((size_t)kMq * 128 * 4);
  float*    dfo  = (float*)alloc((size_t)kMq * kC * 4);        // attn proj, later deform out
  float*    ffh  = (float*)alloc((size_t)kMq * kFF * 4);
  _Float16* kfh  = (_Float16*)alloc((size_t)kMv * kC * 2);
  _Float16* valh = (_Float16*)alloc((size_t)kMv * kC * 2);

  // --- self attention block ---
  make_xq<<<kMq, 256, 0, stream>>>(query, xq);
  gemm_f(xq, W_in, b_in, qkv, kMq, 3 * kC, kC, 0, stream);
  {
    dim3 g((kPQ + 15) / 16, kB * kNH);
    flash_attn<<<g, 32, 0, stream>>>(qkv, attn);
  }
  gemm_f(attn, W_out, b_out, dfo, kMq, kC, kC, 0, stream);
  add_ln<<<kMq, 256, 0, stream>>>(dfo, xq, g1, be1, q1);

  // --- deformable attention block ---
  make_kf<<<(unsigned)kMv, 256, 0, stream>>>(key, kfh);
  gemm_h(kfh, W_v, b_v, valh, kMv, kC, kC, stream);
  gemm_f(q1, W_off, b_off, offb, kMq, 256, kC, 0, stream);
  gemm_f(q1, W_aw, b_aw, awb, kMq, 128, kC, 0, stream);
  aw_softmax<<<(kMq * kNH + 255) / 256, 256, 0, stream>>>(awb);
  ms_deform_k<<<kMq, 256, 0, stream>>>(valh, offb, awb, refp, dfo);
  gemm_f(dfo, W_o, b_o, attn, kMq, kC, kC, 0, stream);
  add_ln<<<kMq, 256, 0, stream>>>(attn, q1, g2, be2, q2);

  // --- FFN block ---
  gemm_f(q2, W1, b1, ffh, kMq, kFF, kC, 1, stream);
  gemm_f(ffh, W2, b2, qkv, kMq, kC, kFF, 0, stream);
  add_ln<<<kMq, 256, 0, stream>>>(qkv, q2, g3, be3, xq);

  make_out<<<kMq, 256, 0, stream>>>(xq, (float*)d_out);
}
